// LightGCL_20779051778134
// MI455X (gfx1250) — compile-verified, matrix operand order checked by reading, not run on hardware
//
#include <hip/hip_runtime.h>
#include <hip/hip_fp16.h>
#include <math.h>

#define U_N 60000
#define I_N 30000
#define E_N 1500000
#define D_N 64
#define Q_N 5
#define B_N 2048
#define TEMP_F 0.2f
#define KEEP_F 0.8f
#define CL_W_F 0.2f
#define L2_W_F 1e-5f

typedef __attribute__((ext_vector_type(16))) _Float16 v16h;
typedef __attribute__((ext_vector_type(8)))  _Float16 v8h;
typedef __attribute__((ext_vector_type(8)))  float    v8f;

// TDM availability: device pass on gfx1250 with the builtin present.
#if defined(__gfx1250__) && defined(__has_builtin)
#if __has_builtin(__builtin_amdgcn_tensor_load_to_lds)
#define HAVE_TDM 1
#endif
#endif
#ifndef HAVE_TDM
#define HAVE_TDM 0
#endif

// ---------------- elementwise helpers ----------------

__global__ void k_zero(float* __restrict__ p, int n) {
    int i = blockIdx.x * blockDim.x + threadIdx.x;
    if (i < n) p[i] = 0.0f;
}

__global__ void k_copy3(const float* __restrict__ src, float* __restrict__ a,
                        float* __restrict__ b, float* __restrict__ c, int n) {
    int i = blockIdx.x * blockDim.x + threadIdx.x;
    if (i < n) { float v = src[i]; a[i] = v; b[i] = v; c[i] = v; }
}

__global__ void k_accum(float* __restrict__ a, const float* __restrict__ b, int n) {
    int i = blockIdx.x * blockDim.x + threadIdx.x;
    if (i < n) a[i] += b[i];
}

__global__ void k_cast(_Float16* __restrict__ h, const float* __restrict__ f, int n) {
    int i = blockIdx.x * blockDim.x + threadIdx.x;
    if (i < n) h[i] = (_Float16)f[i];
}

__global__ void k_gather_cast(_Float16* __restrict__ out, const float* __restrict__ emb,
                              const int* __restrict__ idx, int n) {
    int i = blockIdx.x * blockDim.x + threadIdx.x;
    if (i < n) {
        int b = i / D_N, d = i % D_N;
        out[i] = (_Float16)emb[(size_t)idx[b] * D_N + d];
    }
}

// ---------------- sparse propagation (edge-parallel, atomic scatter) ----------------

__global__ void k_spmm(float* __restrict__ out, const float* __restrict__ src,
                       const int* __restrict__ scat, const int* __restrict__ gath,
                       const float* __restrict__ vals, unsigned layer) {
    int e = blockIdx.x;           // one block per edge
    int d = threadIdx.x;          // 64 threads = one embedding row
    // deterministic hash-based dropout (edge, layer)
    unsigned x = (unsigned)e * 2654435761u ^ (layer * 0x9E3779B9u) ^ 42u;
    x ^= x >> 16; x *= 0x85ebca6bu; x ^= x >> 13; x *= 0xc2b2ae35u; x ^= x >> 16;
    float r = (float)(x >> 8) * (1.0f / 16777216.0f);
    if (r < KEEP_F) {
        float dv = vals[e] * (1.0f / KEEP_F);
        int s = scat[e], g = gath[e];
        atomicAdd(&out[(size_t)s * D_N + d], dv * src[(size_t)g * D_N + d]);
    }
}

// ---------------- low-rank SVD path (Q=5, tiny) ----------------

__global__ void k_svdT(float* __restrict__ tmp, const float* __restrict__ mat,
                       const float* __restrict__ emb, int nrows) {
    int q = threadIdx.x / D_N, d = threadIdx.x % D_N;
    float s = 0.0f;
    for (int i = 0; i < nrows; ++i) s += mat[(size_t)i * Q_N + q] * emb[(size_t)i * D_N + d];
    tmp[q * D_N + d] = s;
}

__global__ void k_svdA(float* __restrict__ outsum, const float* __restrict__ mat,
                       const float* __restrict__ sigma, const float* __restrict__ tmp,
                       int nrows) {
    int i = blockIdx.x * blockDim.x + threadIdx.x;
    if (i >= nrows * D_N) return;
    int u = i / D_N, d = i % D_N;
    float s = 0.0f;
#pragma unroll
    for (int q = 0; q < Q_N; ++q) s += mat[(size_t)u * Q_N + q] * sigma[q] * tmp[q * D_N + d];
    outsum[i] += s;
}

// ---------------- WMMA fragment load: two aligned 16B chunks per lane ----------------
// 16-bit A/B layout (ISA 7.12.2): lanes 0-15 hold K {kb..kb+7, kb+16..kb+23} with kb=0,
// lanes 16-31 the same with kb=8.  Both chunks are 16B-contiguous -> b128 loads.

__device__ inline v16h load_frag(const _Float16* base, int kb) {
    v8h lo = *(const v8h*)(base + kb);
    v8h hi = *(const v8h*)(base + kb + 16);
    v16h r;
#pragma unroll
    for (int j = 0; j < 8; ++j) { r[j] = lo[j]; r[j + 8] = hi[j]; }
    return r;
}

// ---------------- Tensor Data Mover: DMA one 16x64-half tile (2KB) global -> LDS --------

#if HAVE_TDM
typedef unsigned int u32x4 __attribute__((ext_vector_type(4)));
typedef int          i32x4 __attribute__((ext_vector_type(4)));
typedef int          i32x8 __attribute__((ext_vector_type(8)));

__device__ inline void tdm_load_tile(_Float16* lds_dst, const _Float16* gsrc) {
    // Low 32 bits of a generic pointer into LDS aperture are the LDS byte offset.
    unsigned lds_off = (unsigned)(size_t)lds_dst;
    unsigned long long ga = (unsigned long long)(size_t)gsrc;
    // D# group 0: count=1 (valid user descriptor), lds_addr, global_addr, type=2
    u32x4 g0;
    g0[0] = 1u;
    g0[1] = lds_off;
    g0[2] = (unsigned)ga;
    g0[3] = (unsigned)((ga >> 32) & 0x01FFFFFFu) | (2u << 30);
    // D# group 1: data_size=1 (2B), tensor 64x16 (tile-local), tile 64x16, stride 64
    i32x8 g1;
    g1[0] = 1 << 16;        // workgroup_mask=0, data_size[17:16]=1 (2 bytes)
    g1[1] = D_N << 16;      // tensor_dim0[79:48] low half = 64
    g1[2] = 16 << 16;       // tensor_dim0 hi = 0 ; tensor_dim1[111:80] low half = 16
    g1[3] = D_N << 16;      // tensor_dim1 hi = 0 ; tile_dim0[127:112] = 64
    g1[4] = 16;             // tile_dim1[143:128] = 16 ; tile_dim2 = 0
    g1[5] = D_N;            // tensor_dim0_stride[207:160] low 32 = 64
    g1[6] = 0;
    g1[7] = 0;
    i32x4 gz; gz[0] = 0; gz[1] = 0; gz[2] = 0; gz[3] = 0;
#if __clang_major__ >= 23
    i32x8 gz8; gz8[0]=0; gz8[1]=0; gz8[2]=0; gz8[3]=0; gz8[4]=0; gz8[5]=0; gz8[6]=0; gz8[7]=0;
    __builtin_amdgcn_tensor_load_to_lds(g0, g1, gz, gz, gz8, 0);
#else
    __builtin_amdgcn_tensor_load_to_lds(g0, g1, gz, gz, 0);
#endif
}
#endif

// ---------------- WMMA score kernel: [B,64] x [N,64]^T, per-row max / sumexp ----------------
// 256 threads = 8 waves; each wave owns one 16-row M-tile; all waves share one 16-col
// B-tile staged in LDS (TDM double-buffered).  MODE 0: row max.  MODE 1: sum exp(.-mu).

template <int MODE>
__global__ __launch_bounds__(256) void k_score(const _Float16* __restrict__ Ag,
                                               const _Float16* __restrict__ Bm, int N,
                                               const float* __restrict__ mu_ptr,
                                               float* __restrict__ row_out) {
    __shared__ __align__(16) _Float16 bbuf[2][16 * D_N];
    const int tid  = threadIdx.x;
    const int wave = tid >> 5;
    const int lane = tid & 31;
    const int half = lane >> 4;
    const int l16  = lane & 15;
    const int kb   = half * 8;
    const int m0   = (blockIdx.x * 8 + wave) * 16;

    const _Float16* ar = Ag + (size_t)(m0 + l16) * D_N;
    v16h a0 = load_frag(ar, kb);        // K = 0..31
    v16h a1 = load_frag(ar + 32, kb);   // K = 32..63

    const float mu = MODE ? mu_ptr[0] : 0.0f;
    float acc[8];
#pragma unroll
    for (int r = 0; r < 8; ++r) acc[r] = MODE ? 0.0f : -3.0e38f;

    const int T = N / 16;
#if HAVE_TDM
    if (tid < 32) tdm_load_tile(bbuf[0], Bm);            // prefetch tile 0
    for (int t = 0; t < T; ++t) {
        const _Float16* bt = bbuf[t & 1];
        if (tid < 32) __builtin_amdgcn_s_wait_tensorcnt(0);
        __syncthreads();                                 // tile t visible; tile t-1 readers done
        if (tid < 32 && (t + 1) < T)
            tdm_load_tile((_Float16*)bbuf[(t + 1) & 1], Bm + (size_t)(t + 1) * 16 * D_N);
#else
    for (int t = 0; t < T; ++t) {
        const _Float16* bt = bbuf[0];
        __syncthreads();                                 // prior readers done
        if (tid < 128)
            ((v8h*)bbuf[0])[tid] = ((const v8h*)(Bm + (size_t)t * 16 * D_N))[tid];
        __syncthreads();
#endif
        const _Float16* br = bt + l16 * D_N;
        v16h b0 = load_frag(br, kb);
        v16h b1 = load_frag(br + 32, kb);
        v8f c = {};
        c = __builtin_amdgcn_wmma_f32_16x16x32_f16(false, a0, false, b0, (short)0, c, false, false);
        c = __builtin_amdgcn_wmma_f32_16x16x32_f16(false, a1, false, b1, (short)0, c, false, false);
#pragma unroll
        for (int r = 0; r < 8; ++r) {
            float v = c[r] * (1.0f / TEMP_F);
            if (MODE) acc[r] += __expf(v - mu);
            else      acc[r] = fmaxf(acc[r], v);
        }
    }
    // reduce across the 16 lanes sharing each output row (xor masks stay in-group)
#pragma unroll
    for (int m = 1; m < 16; m <<= 1) {
#pragma unroll
        for (int r = 0; r < 8; ++r) {
            float o = __shfl_xor(acc[r], m, 32);
            acc[r] = MODE ? (acc[r] + o) : fmaxf(acc[r], o);
        }
    }
    if (l16 == 0) {
#pragma unroll
        for (int r = 0; r < 8; ++r) row_out[m0 + r + 8 * half] = acc[r];
    }
}

// ---------------- reductions ----------------

__global__ void k_rowmax_reduce(const float* __restrict__ rm, int n, float* __restrict__ out) {
    __shared__ float sm[256];
    float m = -3.0e38f;
    for (int i = threadIdx.x; i < n; i += 256) m = fmaxf(m, rm[i]);
    sm[threadIdx.x] = m;
    __syncthreads();
    for (int s = 128; s > 0; s >>= 1) {
        if (threadIdx.x < s) sm[threadIdx.x] = fmaxf(sm[threadIdx.x], sm[threadIdx.x + s]);
        __syncthreads();
    }
    if (threadIdx.x == 0) out[0] = sm[0];
}

__device__ inline float dot64(const float* __restrict__ a, const float* __restrict__ b) {
    float s = 0.0f;
#pragma unroll
    for (int d = 0; d < 64; ++d) s += a[d] * b[d];
    return s;
}

__global__ void k_pos_bpr(const float* __restrict__ us_sum, const float* __restrict__ u_sum,
                          const float* __restrict__ is_sum, const float* __restrict__ i_sum,
                          const int* __restrict__ user, const int* __restrict__ item,
                          const int* __restrict__ pos, const int* __restrict__ neg,
                          float* __restrict__ scal) {
    int b = blockIdx.x * blockDim.x + threadIdx.x;
    if (b >= B_N) return;
    int u = user[b], it = item[b], p = pos[b], n = neg[b];
    float up = dot64(&us_sum[(size_t)u * D_N], &u_sum[(size_t)u * D_N]) * (1.0f / TEMP_F);
    float ip = dot64(&is_sum[(size_t)it * D_N], &i_sum[(size_t)it * D_N]) * (1.0f / TEMP_F);
    up = fminf(fmaxf(up, -5.0f), 5.0f);
    ip = fminf(fmaxf(ip, -5.0f), 5.0f);
    float x = dot64(&u_sum[(size_t)u * D_N], &i_sum[(size_t)p * D_N]) -
              dot64(&u_sum[(size_t)u * D_N], &i_sum[(size_t)n * D_N]);
    float bpr = logf(1.0f + __expf(-x));   // -log(sigmoid(x))
    atomicAdd(&scal[2], up);
    atomicAdd(&scal[3], ip);
    atomicAdd(&scal[4], bpr);
}

__global__ void k_reg(const float* __restrict__ ue0, const float* __restrict__ ie0,
                      float* __restrict__ scal) {
    __shared__ float sm[256];
    const int total = U_N * D_N + I_N * D_N;
    float s = 0.0f;
    for (int i = blockIdx.x * blockDim.x + threadIdx.x; i < total; i += gridDim.x * blockDim.x) {
        float v = (i < U_N * D_N) ? ue0[i] : ie0[i - U_N * D_N];
        s += v * v;
    }
    sm[threadIdx.x] = s;
    __syncthreads();
    for (int k = 128; k > 0; k >>= 1) {
        if (threadIdx.x < k) sm[threadIdx.x] += sm[threadIdx.x + k];
        __syncthreads();
    }
    if (threadIdx.x == 0) atomicAdd(&scal[5], sm[0]);
}

__global__ void k_finalize(const float* __restrict__ rsum_u, const float* __restrict__ rsum_i,
                           const float* __restrict__ scal, float* __restrict__ out) {
    __shared__ float su[256], si[256];
    float a = 0.0f, b = 0.0f;
    for (int i = threadIdx.x; i < B_N; i += 256) {
        a += logf(rsum_u[i] + 1e-8f);
        b += logf(rsum_i[i] + 1e-8f);
    }
    su[threadIdx.x] = a;
    si[threadIdx.x] = b;
    __syncthreads();
    for (int k = 128; k > 0; k >>= 1) {
        if (threadIdx.x < k) { su[threadIdx.x] += su[threadIdx.x + k]; si[threadIdx.x] += si[threadIdx.x + k]; }
        __syncthreads();
    }
    if (threadIdx.x == 0) {
        float user_neg = su[0] / (float)B_N + scal[0];
        float item_neg = si[0] / (float)B_N + scal[1];
        float user_pos = scal[2] / (float)B_N;
        float item_pos = scal[3] / (float)B_N;
        float bpr = scal[4] / (float)B_N;
        float reg = scal[5];
        float contrastive = -(user_pos + item_pos) + user_neg + item_neg;
        float total = bpr + contrastive * CL_W_F + reg * L2_W_F;
        out[0] = total; out[1] = contrastive; out[2] = bpr; out[3] = reg;
    }
}

// ---------------- host orchestration ----------------

extern "C" void kernel_launch(void* const* d_in, const int* in_sizes, int n_in,
                              void* d_out, int out_size, void* d_ws, size_t ws_size,
                              hipStream_t stream) {
    (void)in_sizes; (void)n_in; (void)out_size; (void)ws_size;
    const int*   rows = (const int*)d_in[0];
    const int*   cols = (const int*)d_in[1];
    const float* vals = (const float*)d_in[2];
    const float* ue0  = (const float*)d_in[3];
    const float* ie0  = (const float*)d_in[4];
    const float* umat = (const float*)d_in[5];
    const float* vmat = (const float*)d_in[6];
    const float* sig  = (const float*)d_in[7];
    const int* user = (const int*)d_in[8];
    const int* item = (const int*)d_in[9];
    const int* pos  = (const int*)d_in[10];
    const int* neg  = (const int*)d_in[11];
    float* out = (float*)d_out;

    const size_t UD = (size_t)U_N * D_N, ID = (size_t)I_N * D_N, BD = (size_t)B_N * D_N;
    float* w = (float*)d_ws;
    size_t off = 0;
    auto alloc = [&](size_t n) { float* p = w + off; off += n; return p; };
    float* ueA = alloc(UD);  float* ueB = alloc(UD);
    float* ieA = alloc(ID);  float* ieB = alloc(ID);
    float* u_sum = alloc(UD);  float* i_sum = alloc(ID);
    float* us_sum = alloc(UD); float* is_sum = alloc(ID);
    float* tmpV = alloc(Q_N * D_N); float* tmpU = alloc(Q_N * D_N);
    float* rmax_u = alloc(B_N); float* rsum_u = alloc(B_N);
    float* rmax_i = alloc(B_N); float* rsum_i = alloc(B_N);
    float* scal = alloc(16);
    _Float16* h = (_Float16*)(w + off);
    size_t hoff = 0;
    auto halloc = [&](size_t n) { _Float16* p = h + hoff; hoff += n; return p; };
    _Float16* u_sum_h = halloc(UD);
    _Float16* i_sum_h = halloc(ID);
    _Float16* usg_h   = halloc(BD);
    _Float16* isg_h   = halloc(BD);

    const int T = 256;
    auto nb = [&](size_t n) { return (unsigned)((n + T - 1) / T); };

    k_zero<<<1, 32, 0, stream>>>(scal, 16);
    k_copy3<<<nb(UD), T, 0, stream>>>(ue0, ueA, u_sum, us_sum, (int)UD);
    k_copy3<<<nb(ID), T, 0, stream>>>(ie0, ieA, i_sum, is_sum, (int)ID);

    float* ue_old = ueA; float* ue_new = ueB;
    float* ie_old = ieA; float* ie_new = ieB;
    for (unsigned layer = 0; layer < 2; ++layer) {
        k_zero<<<nb(UD), T, 0, stream>>>(ue_new, (int)UD);
        k_zero<<<nb(ID), T, 0, stream>>>(ie_new, (int)ID);
        k_spmm<<<E_N, D_N, 0, stream>>>(ue_new, ie_old, rows, cols, vals, layer);
        k_spmm<<<E_N, D_N, 0, stream>>>(ie_new, ue_old, cols, rows, vals, layer);
        k_svdT<<<1, Q_N * D_N, 0, stream>>>(tmpV, vmat, ie_old, I_N);
        k_svdT<<<1, Q_N * D_N, 0, stream>>>(tmpU, umat, ue_old, U_N);
        k_svdA<<<nb(UD), T, 0, stream>>>(us_sum, umat, sig, tmpV, U_N);
        k_svdA<<<nb(ID), T, 0, stream>>>(is_sum, vmat, sig, tmpU, I_N);
        k_accum<<<nb(UD), T, 0, stream>>>(u_sum, ue_new, (int)UD);
        k_accum<<<nb(ID), T, 0, stream>>>(i_sum, ie_new, (int)ID);
        float* t;
        t = ue_old; ue_old = ue_new; ue_new = t;
        t = ie_old; ie_old = ie_new; ie_new = t;
    }

    k_cast<<<nb(UD), T, 0, stream>>>(u_sum_h, u_sum, (int)UD);
    k_cast<<<nb(ID), T, 0, stream>>>(i_sum_h, i_sum, (int)ID);
    k_gather_cast<<<nb(BD), T, 0, stream>>>(usg_h, us_sum, user, (int)BD);
    k_gather_cast<<<nb(BD), T, 0, stream>>>(isg_h, is_sum, item, (int)BD);

    // InfoNCE negatives: WMMA pass 1 (row max) -> global mu -> WMMA pass 2 (sum exp)
    k_score<0><<<B_N / 128, 256, 0, stream>>>(usg_h, u_sum_h, U_N, scal + 0, rmax_u);
    k_score<0><<<B_N / 128, 256, 0, stream>>>(isg_h, i_sum_h, I_N, scal + 1, rmax_i);
    k_rowmax_reduce<<<1, 256, 0, stream>>>(rmax_u, B_N, scal + 0);
    k_rowmax_reduce<<<1, 256, 0, stream>>>(rmax_i, B_N, scal + 1);
    k_score<1><<<B_N / 128, 256, 0, stream>>>(usg_h, u_sum_h, U_N, scal + 0, rsum_u);
    k_score<1><<<B_N / 128, 256, 0, stream>>>(isg_h, i_sum_h, I_N, scal + 1, rsum_i);

    k_pos_bpr<<<nb(B_N), T, 0, stream>>>(us_sum, u_sum, is_sum, i_sum, user, item, pos, neg, scal);
    k_reg<<<512, T, 0, stream>>>(ue0, ie0, scal);
    k_finalize<<<1, 256, 0, stream>>>(rsum_u, rsum_i, scal, out);
}
